// SCGSFM_24713241821853
// MI455X (gfx1250) — matrix operand order repaired and checked by reading, not compiled, not run on hardware
//
#include <hip/hip_runtime.h>
#include <hip/hip_bf16.h>

// ---------------------------------------------------------------------------
// SCGSFM dual-attention block on gfx1250 (MI455X).
// All GEMMs: A row-major [M][K], B given TRANSPOSED (Bt[n][k], K-contiguous).
// Both operand tiles stage via global_load_async_to_lds_b128 (double-buffered),
// fragments via ds_load_b128, math via v_wmma_f32_16x16x32_f16.
// Algebra: S = X1@G feeds both softmaxes (z_c logits are S^T);
//          rowsoftmax(S^T) == colsoftmax(S) == z_c^T;  p_c = X1 @ z_c^T.
// ---------------------------------------------------------------------------

typedef __attribute__((ext_vector_type(16))) _Float16 v16h;
typedef __attribute__((ext_vector_type(8)))  _Float16 v8h;
typedef __attribute__((ext_vector_type(8)))  float    v8f;

#define LDS_STRIDE 40   // 32 K-halves + 8 pad; 80B rows keep 16B alignment

__device__ __forceinline__ void async_copy_b128(unsigned lds_addr, const _Float16* gptr) {
    // VDST = LDS byte address (addr[31:0] of flat shared ptr), VADDR = 64-bit global.
    asm volatile("global_load_async_to_lds_b128 %0, %1, off"
                 :: "v"(lds_addr), "v"(gptr) : "memory");
}

// --------------------------- f32 -> f16 convert ----------------------------
__global__ __launch_bounds__(256)
void f32_to_f16_kernel(const float* __restrict__ src, _Float16* __restrict__ dst, int n) {
    int i = blockIdx.x * 256 + threadIdx.x;
    if (i < n) dst[i] = (_Float16)src[i];
}

// ------------------ f32 [z][R][C] -> f16 transposed [z][C][R] --------------
__global__ __launch_bounds__(256)
void cvt_transpose_kernel(const float* __restrict__ src, _Float16* __restrict__ dst,
                          int R, int C) {
    __shared__ float tile[32][33];
    const int r0 = blockIdx.y * 32, c0 = blockIdx.x * 32;
    const float* s = src + (long long)blockIdx.z * R * C;
    _Float16*    d = dst + (long long)blockIdx.z * R * C;
    const int tx = threadIdx.x, ty = threadIdx.y;          // 32 x 8
    #pragma unroll
    for (int i = 0; i < 4; ++i)
        tile[ty + 8 * i][tx] = s[(long long)(r0 + ty + 8 * i) * C + c0 + tx];
    __syncthreads();
    #pragma unroll
    for (int i = 0; i < 4; ++i)
        d[(long long)(c0 + ty + 8 * i) * R + r0 + tx] = (_Float16)tile[tx][ty + 8 * i];
}

// ------------------------------ WMMA GEMM ----------------------------------
// D = scale*(A @ Bt^T) + bias[row];  ROW: row-major OutT out (+= if ACCUM);
// TR: f16 transposed out Ct[n][m] (contiguous v8h stores).
// M,N multiples of 128; K multiple of 32.
template<typename OutT, bool ROW, bool TR, bool ACCUM>
__global__ __launch_bounds__(256)
void wmma_gemm_kernel(const _Float16* __restrict__ A,
                      const _Float16* __restrict__ Bt,
                      OutT* __restrict__ Crow,
                      _Float16* __restrict__ Ctr,
                      const float* __restrict__ bias,
                      const float* __restrict__ scalep,
                      int K, int lda, int ldbt, int ldc, int ldct,
                      long long strideA, long long strideB,
                      long long strideCrow, long long strideCtr)
{
    __shared__ __attribute__((aligned(16))) _Float16 sA[2][128 * LDS_STRIDE]; // [m][k]
    __shared__ __attribute__((aligned(16))) _Float16 sB[2][128 * LDS_STRIDE]; // [n][k]

    const int tid  = threadIdx.x;
    const int lane = tid & 31;
    const int wv   = tid >> 5;       // wave 0..7
    const int wm   = wv >> 2;        // 0..1 : 64-row slab
    const int wn   = wv & 3;         // 0..3 : 32-col slab
    const int i16  = lane & 15;
    const int hi   = lane >> 4;      // half-wave
    const int kbA  = hi ? 16 : 0;    // A frag: K 0..15 / 16..31
    const int kbB  = hi ? 16 : 0;    // B frag: K 0..15 / 16..31

    const int bM0 = blockIdx.y * 128;
    const int bN0 = blockIdx.x * 128;
    const long long bz = blockIdx.z;

    const _Float16* Ab = A  + bz * strideA;
    const _Float16* Bb = Bt + bz * strideB;

    // per-thread staging geometry: 512 chunks of 8 halves per 128x32 tile
    const int rT = tid >> 2;                 // rows rT and rT+64
    const int kg = (tid & 3) << 3;           // 8-half group within K-step
    const _Float16* gA0 = Ab + (long long)(bM0 + rT) * lda  + kg;
    const _Float16* gA1 = gA0 + (long long)64 * lda;
    const _Float16* gB0 = Bb + (long long)(bN0 + rT) * ldbt + kg;
    const _Float16* gB1 = gB0 + (long long)64 * ldbt;
    unsigned lA0[2], lA1[2], lB0[2], lB1[2];
    #pragma unroll
    for (int b = 0; b < 2; ++b) {
        lA0[b] = (unsigned)(size_t)&sA[b][rT * LDS_STRIDE + kg];
        lA1[b] = (unsigned)(size_t)&sA[b][(rT + 64) * LDS_STRIDE + kg];
        lB0[b] = (unsigned)(size_t)&sB[b][rT * LDS_STRIDE + kg];
        lB1[b] = (unsigned)(size_t)&sB[b][(rT + 64) * LDS_STRIDE + kg];
    }

    auto issue = [&](int it) {
        const long long ko = (long long)it << 5;   // K offset in halves
        const int b = it & 1;
        async_copy_b128(lA0[b], gA0 + ko);
        async_copy_b128(lA1[b], gA1 + ko);
        async_copy_b128(lB0[b], gB0 + ko);
        async_copy_b128(lB1[b], gB1 + ko);
    };

    v8f acc[4][2] = {};
    const int nk = K >> 5;

    issue(0);
    for (int it = 0; it < nk; ++it) {
        if (it + 1 < nk) {
            issue(it + 1);                                   // prefetch next stage
            asm volatile("s_wait_asynccnt 0x4" ::: "memory"); // stage `it` landed
        } else {
            asm volatile("s_wait_asynccnt 0x0" ::: "memory");
        }
        __syncthreads();

        const _Float16* bA = sA[it & 1];
        const _Float16* bB = sB[it & 1];

        v16h af[4], bf[2];
        #pragma unroll
        for (int tm = 0; tm < 4; ++tm) {
            const _Float16* p = &bA[(wm * 64 + tm * 16 + i16) * LDS_STRIDE];
            v8h lo = *reinterpret_cast<const v8h*>(p + kbA);
            v8h h8 = *reinterpret_cast<const v8h*>(p + kbA + 8);
            v16h f;
            #pragma unroll
            for (int j = 0; j < 8; ++j) { f[j] = lo[j]; f[j + 8] = h8[j]; }
            af[tm] = f;
        }
        #pragma unroll
        for (int tn = 0; tn < 2; ++tn) {
            const _Float16* p = &bB[(wn * 32 + tn * 16 + i16) * LDS_STRIDE];
            v8h lo = *reinterpret_cast<const v8h*>(p + kbB);
            v8h h8 = *reinterpret_cast<const v8h*>(p + kbB + 8);
            v16h f;
            #pragma unroll
            for (int j = 0; j < 8; ++j) { f[j] = lo[j]; f[j + 8] = h8[j]; }
            bf[tn] = f;
        }

        #pragma unroll
        for (int tm = 0; tm < 4; ++tm)
            #pragma unroll
            for (int tn = 0; tn < 2; ++tn)
                acc[tm][tn] = __builtin_amdgcn_wmma_f32_16x16x32_f16(
                    false, af[tm], false, bf[tn], (short)0, acc[tm][tn], false, false);

        __syncthreads();
    }

    const float s = scalep ? scalep[0] : 1.0f;

    // C/D layout: VGPR r, lane<16 -> (M=r, N=lane); lane>=16 -> (M=r+8, N=lane-16)
    #pragma unroll
    for (int tm = 0; tm < 4; ++tm) {
        #pragma unroll
        for (int tn = 0; tn < 2; ++tn) {
            const int row0 = bM0 + wm * 64 + tm * 16 + hi * 8;
            const int col  = bN0 + wn * 32 + tn * 16 + i16;
            float vals[8];
            #pragma unroll
            for (int r = 0; r < 8; ++r) {
                float v = s * acc[tm][tn][r];
                if (bias) v += bias[row0 + r];
                vals[r] = v;
            }
            if constexpr (ROW) {
                OutT* Cb = Crow + bz * strideCrow;
                #pragma unroll
                for (int r = 0; r < 8; ++r) {
                    long long idx = (long long)(row0 + r) * ldc + col;
                    if constexpr (ACCUM) Cb[idx] = Cb[idx] + (OutT)vals[r];
                    else                 Cb[idx] = (OutT)vals[r];
                }
            }
            if constexpr (TR) {      // transposed f16: contiguous 16B store
                _Float16* Cb = Ctr + bz * strideCtr;
                v8h h;
                #pragma unroll
                for (int r = 0; r < 8; ++r) h[r] = (_Float16)vals[r];
                *reinterpret_cast<v8h*>(Cb + (long long)col * ldct + row0) = h;
            }
        }
    }
}

// -------------------------- row softmax (f16 io) ---------------------------
__global__ __launch_bounds__(256)
void softmax_rows_kernel(const _Float16* __restrict__ S, _Float16* __restrict__ Z, int N)
{
    const long long base = ((long long)blockIdx.y * N + blockIdx.x) * N;
    const _Float16* p = S + base;
    _Float16*       q = Z + base;
    __shared__ float red[256];
    const int tid = threadIdx.x;

    float vmax = -3.4e38f;
    for (int j = tid; j < N; j += 256) vmax = fmaxf(vmax, (float)p[j]);
    red[tid] = vmax; __syncthreads();
    for (int s = 128; s > 0; s >>= 1) { if (tid < s) red[tid] = fmaxf(red[tid], red[tid + s]); __syncthreads(); }
    const float m = red[0];
    __syncthreads();

    float sum = 0.0f;
    for (int j = tid; j < N; j += 256) sum += __expf((float)p[j] - m);
    red[tid] = sum; __syncthreads();
    for (int s = 128; s > 0; s >>= 1) { if (tid < s) red[tid] += red[tid + s]; __syncthreads(); }
    const float inv = 1.0f / red[0];

    for (int j = tid; j < N; j += 256) q[j] = (_Float16)(__expf((float)p[j] - m) * inv);
}

// ------------------------------- launcher ----------------------------------
extern "C" void kernel_launch(void* const* d_in, const int* in_sizes, int n_in,
                              void* d_out, int out_size, void* d_ws, size_t ws_size,
                              hipStream_t stream) {
    (void)in_sizes; (void)n_in; (void)out_size; (void)ws_size;

    const float* x     = (const float*)d_in[0];   // [B,C,H,W]
    const float* w1    = (const float*)d_in[1];   // [N,C]
    const float* b1    = (const float*)d_in[2];
    const float* wq    = (const float*)d_in[3];   // [N,N]
    const float* bq    = (const float*)d_in[4];
    const float* wv    = (const float*)d_in[5];   // [N,N]
    const float* bv    = (const float*)d_in[6];
    const float* w2    = (const float*)d_in[7];   // [C,N]
    const float* b2    = (const float*)d_in[8];
    const float* alpha = (const float*)d_in[9];
    const float* beta  = (const float*)d_in[10];
    float* out = (float*)d_out;                   // [B,C,H,W] f32

    constexpr int  Bn = 8, Cc = 256, Nn = 1024;
    constexpr long long NN = (long long)Nn * Nn;
    constexpr long long CN = (long long)Cc * Nn;

    char* ws = (char*)d_ws;
    size_t off = 0;
    auto carve = [&](size_t elems) -> _Float16* {
        _Float16* p = (_Float16*)(ws + off);
        off = (off + elems * sizeof(_Float16) + 255) & ~(size_t)255;
        return p;
    };
    _Float16* Xt   = carve((size_t)Bn * CN);   // x^T per batch: [N][C]
    _Float16* W1h  = carve((size_t)Nn * Cc);
    _Float16* Wqh  = carve((size_t)Nn * Nn);
    _Float16* Wvh  = carve((size_t)Nn * Nn);
    _Float16* W2h  = carve((size_t)Cc * Nn);
    _Float16* X1   = carve((size_t)Bn * NN);   // row-major (A operand)
    _Float16* X1t  = carve((size_t)Bn * NN);   // transposed (B operand)
    _Float16* Gt   = carve((size_t)Bn * NN);
    _Float16* Rt   = carve((size_t)Bn * NN);
    _Float16* S    = carve((size_t)Bn * NN);   // reused as Ps^T
    _Float16* St   = carve((size_t)Bn * NN);   // reused as Pc^T
    _Float16* Zs   = carve((size_t)Bn * NN);
    _Float16* ZctT = carve((size_t)Bn * NN);

    auto cvt = [&](const float* s, _Float16* d, int n) {
        f32_to_f16_kernel<<<(n + 255) / 256, 256, 0, stream>>>(s, d, n);
    };
    // x [B][C][N] -> Xt [B][N][C]
    cvt_transpose_kernel<<<dim3(Nn / 32, Cc / 32, Bn), dim3(32, 8), 0, stream>>>(x, Xt, Cc, Nn);
    cvt(w1, W1h, Nn * Cc);
    cvt(wq, Wqh, Nn * Nn);
    cvt(wv, Wvh, Nn * Nn);
    cvt(w2, W2h, Cc * Nn);

    const dim3 blk(256);
    const dim3 gNN(Nn / 128, Nn / 128, Bn);   // 1024x1024 outputs
    const dim3 gCN(Nn / 128, Cc / 128, Bn);   // 256x1024 outputs

    // 1) X1 = W1 @ X + b1  -> row + transposed   (K = 256)
    wmma_gemm_kernel<_Float16, true, true, false><<<gNN, blk, 0, stream>>>(
        W1h, Xt, X1, X1t, b1, nullptr, Cc, Cc, Cc, Nn, Nn, 0LL, CN, NN, NN);
    // 2) G^T = (Wq @ X1 + bq)^T
    wmma_gemm_kernel<_Float16, false, true, false><<<gNN, blk, 0, stream>>>(
        Wqh, X1t, (_Float16*)nullptr, Gt, bq, nullptr, Nn, Nn, Nn, 0, Nn, 0LL, NN, 0LL, NN);
    // 3) R^T = (Wv @ X1 + bv)^T
    wmma_gemm_kernel<_Float16, false, true, false><<<gNN, blk, 0, stream>>>(
        Wvh, X1t, (_Float16*)nullptr, Rt, bv, nullptr, Nn, Nn, Nn, 0, Nn, 0LL, NN, 0LL, NN);
    // 4) S = X1 @ G  -> row (S) + transposed (S^T)
    wmma_gemm_kernel<_Float16, true, true, false><<<gNN, blk, 0, stream>>>(
        X1, Gt, S, St, nullptr, nullptr, Nn, Nn, Nn, Nn, Nn, NN, NN, NN, NN);

    // 5) Zs = rowsoftmax(S);  Zc^T = rowsoftmax(S^T)  (== column softmax of S)
    softmax_rows_kernel<<<dim3(Nn, Bn), blk, 0, stream>>>(S,  Zs,   Nn);
    softmax_rows_kernel<<<dim3(Nn, Bn), blk, 0, stream>>>(St, ZctT, Nn);

    // 6) Ps^T = (Zs @ R)^T  (into S buffer);  Pc^T = (X1 @ Zc^T)^T (into St buffer)
    _Float16* PsT = S;
    _Float16* PcT = St;
    wmma_gemm_kernel<_Float16, false, true, false><<<gNN, blk, 0, stream>>>(
        Zs, Rt, (_Float16*)nullptr, PsT, nullptr, nullptr, Nn, Nn, Nn, 0, Nn, NN, NN, 0LL, NN);
    wmma_gemm_kernel<_Float16, false, true, false><<<gNN, blk, 0, stream>>>(
        X1, ZctT, (_Float16*)nullptr, PcT, nullptr, nullptr, Nn, Nn, Nn, 0, Nn, NN, NN, 0LL, NN);

    // 7) out = alpha * W2@Ps + b2 ;  out += beta * W2@Pc
    wmma_gemm_kernel<float, true, false, false><<<gCN, blk, 0, stream>>>(
        W2h, PsT, out, (_Float16*)nullptr, b2, alpha, Nn, Nn, Nn, Nn, 0, 0LL, NN, CN, 0LL);
    wmma_gemm_kernel<float, true, false, true><<<gCN, blk, 0, stream>>>(
        W2h, PcT, out, (_Float16*)nullptr, nullptr, beta, Nn, Nn, Nn, Nn, 0, 0LL, NN, CN, 0LL);
}
